// HNetEnergyViaEdgeMatching_42271068127504
// MI455X (gfx1250) — 8-bit matrix operand order repaired and checked by reading, compiled, not run on hardware
//
#include <hip/hip_runtime.h>
#include <stdint.h>

typedef __attribute__((ext_vector_type(16))) int          v16i;
typedef __attribute__((ext_vector_type(8)))  float        v8f;
typedef __attribute__((ext_vector_type(4)))  unsigned int v4u;

#define N_PTS   2048
#define N_NODES 1024
#define N_EDGES 8192
#define N_CMP   4096
#define KBYTES  (N_EDGES * 4)
#define KT      (KBYTES / 128)
#define PANEL   2048

#define WS_A_OFF    ((size_t)0)
#define WS_B_OFF    ((size_t)(N_PTS / 16) * KT * PANEL)
#define WS_NULL_OFF (WS_B_OFF + (size_t)(N_CMP / 16) * KT * PANEL)
#define WS_MIN_OFF  (WS_NULL_OFF + (size_t)N_CMP * sizeof(int))


__global__ __launch_bounds__(256)
void init_ws(int* __restrict__ nullcnt, unsigned int* __restrict__ minslot) {
    for (int i = threadIdx.x; i < N_CMP; i += 256) nullcnt[i] = 0;
    if (threadIdx.x == 0) *minslot = 0x7F800000u;
}

__global__ __launch_bounds__(256)
void encode_points(const int* __restrict__ node, const int* __restrict__ eidx,
                   const int* __restrict__ filt, uint8_t* __restrict__ Atile) {
    __shared__ uint32_t stage[4096];
    const int mt  = blockIdx.y;
    const int ktg = blockIdx.x;
    const int tid = threadIdx.x;
    const int f0 = filt[0], f1 = filt[1];

    const int e  = ktg * 256 + tid;
    const int i0 = eidx[2 * e + 0];
    const int i1 = eidx[2 * e + 1];

    const int e32 = tid & 31;
    const int p   = e32 >> 2;
    const int sel = (e32 >> 1) & 1;
    const int bo  = (e32 & 1) * 4;
    const int baseoff = (tid >> 5) * PANEL + (p >> 1) * 512 + (16 * sel) * 16
                      + (p & 1) * 8 + bo;

    #pragma unroll
    for (int m = 0; m < 16; ++m) {
        const int row = mt * 16 + m;
        int t = node[row * N_NODES + i0] * 2 + node[row * N_NODES + i1];
        int c = (t == 0) ? 2 : (t == 1) ? 3 : (t == 2) ? 5 : 9;
        uint32_t oh = (c == f0 || c == f1) ? (0x38u << (8 * t)) : 0u;
        stage[(baseoff + m * 16) >> 2] = oh;
    }
    __syncthreads();
    uint4*       dst = (uint4*)(Atile + ((size_t)mt * KT + (size_t)ktg * 8) * PANEL);
    const uint4* src = (const uint4*)stage;
    #pragma unroll
    for (int c2 = 0; c2 < 4; ++c2) dst[tid + c2 * 256] = src[tid + c2 * 256];
}

__global__ __launch_bounds__(256)
void encode_templates(const float* __restrict__ L, uint8_t* __restrict__ Btile,
                      int* __restrict__ nullcnt) {
    __shared__ uint32_t stage[4096];
    __shared__ int nc[16];
    const int nt  = blockIdx.y;
    const int ktg = blockIdx.x;
    const int tid = threadIdx.x;
    const int lane = tid & 31;
    if (tid < 16) nc[tid] = 0;
    __syncthreads();

    const int e   = ktg * 256 + tid;
    const int e32 = tid & 31;
    const int q   = e32 >> 3;
    const int sel = (e32 >> 1) & 1;
    const int bo  = ((e32 >> 2) & 1) * 8 + (e32 & 1) * 4;
    const int baseoff = (tid >> 5) * PANEL + q * 512 + (16 * sel) * 16 + bo;

    #pragma unroll
    for (int m = 0; m < 16; ++m) {
        const int j = nt * 16 + m;
        int v = (int)L[(size_t)j * N_EDGES + e];
        uint32_t oh = 0u;
        oh |= (v == 2) ? 0x00000038u : 0u;
        oh |= (v == 3) ? 0x00003800u : 0u;
        oh |= (v == 5) ? 0x00380000u : 0u;
        oh |= (v == 9) ? 0x38000000u : 0u;
        stage[(baseoff + m * 16) >> 2] = oh;
        unsigned long long bal = __ballot(v == 0);
        if (lane == 0) atomicAdd(&nc[m], __popcll(bal));
    }
    __syncthreads();
    if (tid < 16) atomicAdd(&nullcnt[nt * 16 + tid], nc[tid]);
    uint4*       dst = (uint4*)(Btile + ((size_t)nt * KT + (size_t)ktg * 8) * PANEL);
    const uint4* src = (const uint4*)stage;
    #pragma unroll
    for (int c2 = 0; c2 < 4; ++c2) dst[tid + c2 * 256] = src[tid + c2 * 256];
}

__global__ __launch_bounds__(256)
void energy_gemm(const uint8_t* __restrict__ Atile, const uint8_t* __restrict__ Btile,
                 const int* __restrict__ nullcnt, float* __restrict__ out,
                 unsigned int* __restrict__ minslot) {
    __shared__ uint8_t lds[2][24576];
    const int tid  = threadIdx.x;
    const int lane = tid & 31;
    const int wave = tid >> 5;
    const int wm   = (wave >> 1) * 32;
    const int wn   = (wave & 1) * 32;
    const int blockM = blockIdx.y * 128;
    const int blockN = blockIdx.x * 64;
    const int mt0 = blockM >> 4;
    const int nt0 = blockN >> 4;
    const int l15 = lane & 15;
    const int sel = lane >> 4;

    uint64_t gA[4]; uint32_t lA[4];
    uint64_t gB[2]; uint32_t lB[2];
    #pragma unroll
    for (int c = 0; c < 4; ++c) {
        int ch = tid + c * 256;
        int pm = ch >> 7;
        gA[c] = (uint64_t)(uintptr_t)(Atile + (size_t)(mt0 + pm) * KT * PANEL
                                            + (ch & 127) * 16);
        lA[c] = (uint32_t)(uintptr_t)(&lds[0][0]) + ch * 16;
    }
    #pragma unroll
    for (int c = 0; c < 2; ++c) {
        int ch = tid + c * 256;
        int pn = ch >> 7;
        gB[c] = (uint64_t)(uintptr_t)(Btile + (size_t)(nt0 + pn) * KT * PANEL
                                            + (ch & 127) * 16);
        lB[c] = (uint32_t)(uintptr_t)(&lds[0][16384]) + ch * 16;
    }

    auto issue = [&](int kt, int buf) {
        const uint32_t lo = buf ? 24576u : 0u;
        const uint64_t ko = (uint64_t)kt * PANEL;
        #pragma unroll
        for (int c = 0; c < 4; ++c)
            asm volatile("global_load_async_to_lds_b128 %0, %1, off"
                         :: "v"(lA[c] + lo), "v"(gA[c] + ko) : "memory");
        #pragma unroll
        for (int c = 0; c < 2; ++c)
            asm volatile("global_load_async_to_lds_b128 %0, %1, off"
                         :: "v"(lB[c] + lo), "v"(gB[c] + ko) : "memory");
    };

    v8f c00 = {}, c01 = {}, c10 = {}, c11 = {};
    const int pa0 = wm >> 4, pa1 = pa0 + 1;
    const int pb0 = wn >> 4, pb1 = pb0 + 1;

    issue(0, 0);
    for (int kt = 0; kt < KT; ++kt) {
        const int cur = kt & 1;
        if (kt + 1 < KT) {
            issue(kt + 1, cur ^ 1);
            asm volatile("s_wait_asynccnt 0x6" ::: "memory");
        } else {
            asm volatile("s_wait_asynccnt 0x0" ::: "memory");
        }
        __syncthreads();

        const uint8_t* A = &lds[cur][0];
        const uint8_t* B = &lds[cur][16384];
        v16i a0, a1, b0, b1;
        #pragma unroll
        for (int i = 0; i < 4; ++i) {
            v4u x0 = *(const v4u*)(A + pa0 * PANEL + i * 512 + lane * 16);
            v4u x1 = *(const v4u*)(A + pa1 * PANEL + i * 512 + lane * 16);
            v4u y0 = *(const v4u*)(B + pb0 * PANEL + i * 512 + lane * 16);
            v4u y1 = *(const v4u*)(B + pb1 * PANEL + i * 512 + lane * 16);
            #pragma unroll
            for (int d = 0; d < 4; ++d) {
                a0[4 * i + d] = (int)x0[d];
                a1[4 * i + d] = (int)x1[d];
                b0[4 * i + d] = (int)y0[d];
                b1[4 * i + d] = (int)y1[d];
            }
        }
        c00 = __builtin_amdgcn_wmma_f32_16x16x128_fp8_fp8(a0, b0, (short)0, c00, false, false);
        c01 = __builtin_amdgcn_wmma_f32_16x16x128_fp8_fp8(a0, b1, (short)0, c01, false, false);
        c10 = __builtin_amdgcn_wmma_f32_16x16x128_fp8_fp8(a1, b0, (short)0, c10, false, false);
        c11 = __builtin_amdgcn_wmma_f32_16x16x128_fp8_fp8(a1, b1, (short)0, c11, false, false);
        __syncthreads();
    }

    const float nc0 = (float)nullcnt[blockN + wn + 0  + l15];
    const float nc1 = (float)nullcnt[blockN + wn + 16 + l15];
    const int col0 = blockN + wn + 0  + l15;
    const int col1 = blockN + wn + 16 + l15;
    float lmin = 3.402823466e38f;
    #pragma unroll
    for (int r = 0; r < 8; ++r) {
        int row0 = blockM + wm + 0  + r + sel * 8;
        int row1 = blockM + wm + 16 + r + sel * 8;
        float e00 = c00[r] + nc0; out[(size_t)row0 * N_CMP + col0] = e00;
        float e01 = c01[r] + nc1; out[(size_t)row0 * N_CMP + col1] = e01;
        float e10 = c10[r] + nc0; out[(size_t)row1 * N_CMP + col0] = e10;
        float e11 = c11[r] + nc1; out[(size_t)row1 * N_CMP + col1] = e11;
        lmin = fminf(lmin, fminf(fminf(e00, e01), fminf(e10, e11)));
    }
    for (int off = 16; off > 0; off >>= 1)
        lmin = fminf(lmin, __shfl_xor(lmin, off, 32));
    if (lane == 0) atomicMin(minslot, __float_as_uint(lmin));
}

__global__ __launch_bounds__(256)
void subtract_min(float* __restrict__ out, const unsigned int* __restrict__ minslot) {
    float m = __uint_as_float(*minslot);
    size_t idx = (size_t)blockIdx.x * 256 + threadIdx.x;
    if (idx < (size_t)N_PTS * N_CMP) out[idx] -= m;
}

extern "C" void kernel_launch(void* const* d_in, const int* in_sizes, int n_in,
                              void* d_out, int out_size, void* d_ws, size_t ws_size,
                              hipStream_t stream) {
    (void)in_sizes; (void)n_in; (void)out_size; (void)ws_size;
    const int*   node = (const int*)d_in[0];
    const float* L    = (const float*)d_in[1];
    const int*   eidx = (const int*)d_in[2];
    const int*   filt = (const int*)d_in[3];
    float* out = (float*)d_out;

    uint8_t* ws = (uint8_t*)d_ws;
    uint8_t*      Atile   = ws + WS_A_OFF;
    uint8_t*      Btile   = ws + WS_B_OFF;
    int*          nullcnt = (int*)(ws + WS_NULL_OFF);
    unsigned int* minslot = (unsigned int*)(ws + WS_MIN_OFF);

    init_ws<<<dim3(1), 256, 0, stream>>>(nullcnt, minslot);
    encode_points<<<dim3(32, N_PTS / 16), 256, 0, stream>>>(node, eidx, filt, Atile);
    encode_templates<<<dim3(32, N_CMP / 16), 256, 0, stream>>>(L, Btile, nullcnt);
    energy_gemm<<<dim3(N_CMP / 64, N_PTS / 128), 256, 0, stream>>>(
        Atile, Btile, nullcnt, out, minslot);
    subtract_min<<<dim3((N_PTS * N_CMP) / 256), 256, 0, stream>>>(out, minslot);
}
